// Attention_12266426597578
// MI455X (gfx1250) — matrix operand order repaired and checked by reading, not compiled, not run on hardware
//
#include <hip/hip_runtime.h>
#include <math.h>

typedef __bf16 v16bf __attribute__((ext_vector_type(16)));
typedef __bf16 v8bf  __attribute__((ext_vector_type(8)));
typedef __bf16 v4bf  __attribute__((ext_vector_type(4)));
typedef float  v8f   __attribute__((ext_vector_type(8)));

static constexpr int S_LEN  = 4096;
static constexpr int NHQ    = 32;
static constexpr int NKV    = 8;
static constexpr int HDIM   = 128;
static constexpr int CHUNKL = 1024;

// ---------------------------------------------------------------------------
// WMMA fragment helpers (CDNA5 wave32, 16x16x32 bf16; ISA 7.12.2 layouts)
// ---------------------------------------------------------------------------
__device__ inline v16bf cat8(v8bf lo, v8bf hi) {
  return __builtin_shufflevector(lo, hi, 0, 1, 2, 3, 4, 5, 6, 7,
                                 8, 9, 10, 11, 12, 13, 14, 15);
}

// A-matrix fragment from row-major [M=16, K] tile: m = lane%16,
// K elems {kb..kb+7, kb+16..kb+23}, kb = (lane<16 ? 0 : 8).
__device__ inline v16bf frag_a(const __bf16* tile, int stride) {
  const int lane = threadIdx.x & 31;
  const __bf16* p = tile + (lane & 15) * stride + ((lane & 16) ? 8 : 0);
  return cat8(*(const v8bf*)p, *(const v8bf*)(p + 16));
}

// B-matrix fragment from row-major [N, K] storage (i.e. B^T of the math B):
// n = lane%16, K elems {kb..kb+7, kb+16..kb+23}, kb = (lane<16 ? 0 : 8).
__device__ inline v16bf frag_b(const __bf16* tile, int stride) {
  const int lane = threadIdx.x & 31;
  const __bf16* p = tile + (lane & 15) * stride + ((lane & 16) ? 8 : 0);
  return cat8(*(const v8bf*)p, *(const v8bf*)(p + 16));
}

__device__ inline v8f wmma_bf16(v16bf a, v16bf b, v8f c) {
  // emits v_wmma_f32_16x16x32_bf16
  return __builtin_amdgcn_wmma_f32_16x16x32_bf16(false, a, false, b,
                                                 (short)0, c, false, false);
}

// Async global->LDS copy of 16B per lane (CDNA5, tracked by ASYNCcnt).
__device__ inline void async_copy_b128(const __bf16* gsrc, __bf16* lds_dst) {
  unsigned lds_off = (unsigned)(uintptr_t)lds_dst;      // addr[31:0] = LDS offset
  unsigned long long ga = (unsigned long long)(uintptr_t)gsrc;
  asm volatile("global_load_async_to_lds_b128 %0, %1, off"
               :: "v"(lds_off), "v"(ga)
               : "memory");
}

__device__ inline void async_wait0() {
  asm volatile("s_wait_asynccnt 0x0" ::: "memory");
}

// ---------------------------------------------------------------------------
// Generic GEMM: C[M,N] = A[M,K] * B[N,K]^T, fp32 in / fp32 out, bf16 WMMA math
// Block: 256 thr (8 waves), tile 128x128, BK=32. Wave = 32 rows x 64 cols.
// N, K compile-time so the epilogue can use immediate-offset stores.
// ---------------------------------------------------------------------------
#define GLD 40  // LDS row stride in bf16 elems (80 B: 16B-aligned, bank-spread)

template <int N, int K>
__global__ __launch_bounds__(256) void gemm_xwT_bf16(
    const float* __restrict__ A, const float* __restrict__ B,
    float* __restrict__ C) {
  __shared__ __bf16 sA[128 * GLD];
  __shared__ __bf16 sB[128 * GLD];

  const int tid  = threadIdx.x;
  const int lane = tid & 31;
  const int wv   = tid >> 5;
  const int wm   = wv >> 1;   // 0..3 -> 32-row slab
  const int wn   = wv & 1;    // 0..1 -> 64-col slab
  const int bm0  = blockIdx.y * 128;
  const int bn0  = blockIdx.x * 128;

  const int lrow = tid >> 1;         // 0..127
  const int lcol = (tid & 1) * 16;   // 0 or 16

  v8f acc[2][4] = {};

  for (int k0 = 0; k0 < K; k0 += 32) {
    {  // stage A tile 128x32 fp32 -> bf16 LDS
      const float* g = A + (size_t)(bm0 + lrow) * K + k0 + lcol;
      alignas(16) __bf16 t[16];
#pragma unroll
      for (int i = 0; i < 4; ++i) {
        float4 f = *(const float4*)(g + i * 4);
        t[i * 4 + 0] = (__bf16)f.x; t[i * 4 + 1] = (__bf16)f.y;
        t[i * 4 + 2] = (__bf16)f.z; t[i * 4 + 3] = (__bf16)f.w;
      }
      *(v8bf*)&sA[lrow * GLD + lcol]     = *(const v8bf*)&t[0];
      *(v8bf*)&sA[lrow * GLD + lcol + 8] = *(const v8bf*)&t[8];
    }
    {  // stage B tile 128x32 fp32 -> bf16 LDS
      const float* g = B + (size_t)(bn0 + lrow) * K + k0 + lcol;
      alignas(16) __bf16 t[16];
#pragma unroll
      for (int i = 0; i < 4; ++i) {
        float4 f = *(const float4*)(g + i * 4);
        t[i * 4 + 0] = (__bf16)f.x; t[i * 4 + 1] = (__bf16)f.y;
        t[i * 4 + 2] = (__bf16)f.z; t[i * 4 + 3] = (__bf16)f.w;
      }
      *(v8bf*)&sB[lrow * GLD + lcol]     = *(const v8bf*)&t[0];
      *(v8bf*)&sB[lrow * GLD + lcol + 8] = *(const v8bf*)&t[8];
    }
    __syncthreads();

    v16bf af[2], bfr[4];
#pragma unroll
    for (int i = 0; i < 2; ++i) af[i] = frag_a(&sA[(wm * 32 + i * 16) * GLD], GLD);
#pragma unroll
    for (int j = 0; j < 4; ++j) bfr[j] = frag_b(&sB[(wn * 64 + j * 16) * GLD], GLD);
#pragma unroll
    for (int i = 0; i < 2; ++i)
#pragma unroll
      for (int j = 0; j < 4; ++j) acc[i][j] = wmma_bf16(af[i], bfr[j], acc[i][j]);
    __syncthreads();
  }

  // C/D layout: vgpr r -> row r (lanes 0-15) / r+8 (lanes 16-31); col = lane%16
  const int ln  = lane & 15;
  const int hi8 = (lane & 16) ? 8 : 0;
#pragma unroll
  for (int i = 0; i < 2; ++i)
#pragma unroll
    for (int j = 0; j < 4; ++j) {
      const int row0 = bm0 + wm * 32 + i * 16 + hi8;
      const int col  = bn0 + wn * 64 + j * 16 + ln;
      float* cp = C + (size_t)row0 * N + col;
#pragma unroll
      for (int r = 0; r < 8; ++r) cp[r * N] = acc[i][j][r];  // imm offsets
    }
}

// ---------------------------------------------------------------------------
// RoPE (interleaved pairs) + RMSNorm over head_dim + long-ctx temp scale,
// fp32 in -> bf16 out. One wave per (s, head) row of 128.
// ---------------------------------------------------------------------------
__global__ __launch_bounds__(256) void rope_rms_cast_kernel(
    const float* __restrict__ in, __bf16* __restrict__ out, int nh, int isq) {
  const int lane = threadIdx.x & 31;
  const int gw = blockIdx.x * (blockDim.x >> 5) + (threadIdx.x >> 5);
  const int s = gw / nh;
  const int h = gw - s * nh;
  const size_t base = ((size_t)s * nh + h) * HDIM;

  float4 v = *(const float4*)(in + base + lane * 4);
  const float fs = (float)s;
  const float kfreq = -0.20503692776104708f;  // -ln(500000)/64
  const float i0 = (float)(lane * 2);
  const float i1 = (float)(lane * 2 + 1);
  float sn0, cs0, sn1, cs1;
  sincosf(fs * expf(kfreq * i0), &sn0, &cs0);
  sincosf(fs * expf(kfreq * i1), &sn1, &cs1);
  float ra0 = v.x * cs0 - v.y * sn0;
  float rb0 = v.x * sn0 + v.y * cs0;
  float ra1 = v.z * cs1 - v.w * sn1;
  float rb1 = v.z * sn1 + v.w * cs1;

  float ss = ra0 * ra0 + rb0 * rb0 + ra1 * ra1 + rb1 * rb1;
#pragma unroll
  for (int m = 16; m >= 1; m >>= 1) ss += __shfl_xor(ss, m, 32);
  float rn = rsqrtf(ss * (1.0f / 128.0f) + 1e-5f);
  if (isq)  // identity for pos < 8192 but computed exactly as the reference
    rn *= logf(floorf((fs + 1.0f) * (1.0f / 8192.0f)) + 1.0f) * 0.1f + 1.0f;

  v4bf o = {(__bf16)(ra0 * rn), (__bf16)(rb0 * rn),
            (__bf16)(ra1 * rn), (__bf16)(rb1 * rn)};
  *(v4bf*)(out + base + lane * 4) = o;
}

__global__ __launch_bounds__(256) void cast_bf16_kernel(
    const float* __restrict__ in, __bf16* __restrict__ out, int n4) {
  int i = blockIdx.x * blockDim.x + threadIdx.x;
  if (i < n4) {
    float4 v = *(const float4*)(in + (size_t)i * 4);
    v4bf o = {(__bf16)v.x, (__bf16)v.y, (__bf16)v.z, (__bf16)v.w};
    *(v4bf*)(out + (size_t)i * 4) = o;
  }
}

// ---------------------------------------------------------------------------
// Chunked-causal flash attention. Grid (qblk 0..7, head 0..31, chunk 0..3).
// 8 waves x 16 q-rows; 64-key tiles; online softmax; bf16 WMMA, fp32 accum.
// K tile staged with CDNA5 async global->LDS; V staged via VGPRs (transpose).
// ---------------------------------------------------------------------------
__global__ __launch_bounds__(256) void attn_chunked_kernel(
    const __bf16* __restrict__ Q,   // [S, 32*128]
    const __bf16* __restrict__ Kt,  // [S, 8*128]
    const __bf16* __restrict__ Vt,  // [S, 8*128]
    float* __restrict__ O) {        // [S, 32*128]
  __shared__ __bf16 sK[64 * 136];   // [key][hd]
  __shared__ __bf16 sVT[128 * 72];  // [hd][key]  (transposed for P*V B-frags)
  __shared__ __bf16 sP[8 * 16 * 72];  // per-wave P tile [16 q][64 key]

  const int tid = threadIdx.x;
  const int lane = tid & 31;
  const int wv = tid >> 5;
  const int ln = lane & 15;
  const int hi8 = (lane & 16) ? 8 : 0;

  const int qblk = blockIdx.x;
  const int h = blockIdx.y;
  const int chunk = blockIdx.z;
  const int kh = h >> 2;  // GQA: 4 q-heads per kv-head
  const int chbase = chunk * CHUNKL;

  // Q fragments for this wave's 16 rows, resident for the whole block
  v16bf qf[4];
  {
    const int qrow = chbase + qblk * 128 + wv * 16 + ln;
    const __bf16* g = Q + (size_t)qrow * (NHQ * HDIM) + h * HDIM +
                      ((lane & 16) ? 8 : 0);
#pragma unroll
    for (int t = 0; t < 4; ++t)
      qf[t] = cat8(*(const v8bf*)(g + t * 32), *(const v8bf*)(g + t * 32 + 16));
  }

  v8f oacc[8] = {};
  float mrow[8], lrow[8];
#pragma unroll
  for (int r = 0; r < 8; ++r) { mrow[r] = -1e30f; lrow[r] = 0.0f; }

  __bf16* sPw = &sP[wv * 16 * 72];
  const int jmax = (qblk + 1) * 2;  // block-causal: keys up to (qblk+1)*128

  for (int j = 0; j < jmax; ++j) {
    const int kb = j * 64;
    {  // stage K [64x128] async, V^T [128x64] via VGPR transpose
      const int r = tid >> 2;          // 0..63
      const int c0 = (tid & 3) * 32;   // 0,32,64,96
      const size_t roff = (size_t)(chbase + kb + r) * (NKV * HDIM) + kh * HDIM + c0;
      const __bf16* gk = Kt + roff;
      const __bf16* gv = Vt + roff;
#pragma unroll
      for (int i = 0; i < 4; ++i)
        async_copy_b128(gk + i * 8, &sK[r * 136 + c0 + i * 8]);
#pragma unroll
      for (int i = 0; i < 4; ++i) {
        v8bf vv = *(const v8bf*)(gv + i * 8);
#pragma unroll
        for (int e = 0; e < 8; ++e) sVT[(c0 + i * 8 + e) * 72 + r] = vv[e];
      }
    }
    async_wait0();
    __syncthreads();

    // scores: 16 q-rows x 64 keys
    v8f sc[4] = {};
#pragma unroll
    for (int t = 0; t < 4; ++t)
#pragma unroll
      for (int f = 0; f < 4; ++f)
        sc[f] = wmma_bf16(qf[t], frag_b(&sK[f * 16 * 136 + t * 32], 136), sc[f]);

    // scale + causal mask (only the diagonal j-tiles actually mask)
    const float inv_sqrt_hd = 0.08838834764831845f;
#pragma unroll
    for (int f = 0; f < 4; ++f) {
      const int kc = kb + f * 16 + ln;
#pragma unroll
      for (int r = 0; r < 8; ++r) {
        const int qr = qblk * 128 + wv * 16 + r + hi8;
        float xv = sc[f][r] * inv_sqrt_hd;
        sc[f][r] = (kc > qr) ? -1e30f : xv;
      }
    }

    // online softmax (row stats replicated across each 16-lane half)
#pragma unroll
    for (int r = 0; r < 8; ++r) {
      float nm = mrow[r];
#pragma unroll
      for (int f = 0; f < 4; ++f) nm = fmaxf(nm, sc[f][r]);
#pragma unroll
      for (int msk = 8; msk >= 1; msk >>= 1)
        nm = fmaxf(nm, __shfl_xor(nm, msk, 32));
      const float alpha = expf(mrow[r] - nm);
      float rs = 0.0f;
#pragma unroll
      for (int f = 0; f < 4; ++f) {
        float p = expf(sc[f][r] - nm);
        rs += p;
        sPw[(r + hi8) * 72 + f * 16 + ln] = (__bf16)p;
      }
#pragma unroll
      for (int msk = 8; msk >= 1; msk >>= 1) rs += __shfl_xor(rs, msk, 32);
      lrow[r] = lrow[r] * alpha + rs;
      mrow[r] = nm;
#pragma unroll
      for (int f8 = 0; f8 < 8; ++f8) oacc[f8][r] *= alpha;
    }

    // sP is private to this wave: drain LDS stores before fragment reads
    asm volatile("s_wait_dscnt 0x0" ::: "memory");

    // O += P * V
#pragma unroll
    for (int t2 = 0; t2 < 2; ++t2) {
      v16bf pf = frag_a(sPw + t2 * 32, 72);
#pragma unroll
      for (int f8 = 0; f8 < 8; ++f8)
        oacc[f8] = wmma_bf16(pf, frag_b(&sVT[f8 * 16 * 72 + t2 * 32], 72),
                             oacc[f8]);
    }
    __syncthreads();
  }

  // normalize + store fp32 [s, h*128+hd]
#pragma unroll
  for (int r = 0; r < 8; ++r) {
    const float inv = 1.0f / lrow[r];
    const int srow = chbase + qblk * 128 + wv * 16 + r + hi8;
#pragma unroll
    for (int f8 = 0; f8 < 8; ++f8)
      O[(size_t)srow * (NHQ * HDIM) + h * HDIM + f8 * 16 + ln] =
          oacc[f8][r] * inv;
  }
}

// ---------------------------------------------------------------------------
extern "C" void kernel_launch(void* const* d_in, const int* in_sizes, int n_in,
                              void* d_out, int out_size, void* d_ws,
                              size_t ws_size, hipStream_t stream) {
  (void)in_sizes; (void)n_in; (void)out_size; (void)ws_size;
  const float* x  = (const float*)d_in[0];
  const float* wq = (const float*)d_in[1];
  const float* wk = (const float*)d_in[2];
  const float* wv = (const float*)d_in[3];
  const float* wo = (const float*)d_in[4];
  float* out = (float*)d_out;

  char* ws = (char*)d_ws;
  float* Qpre = (float*)ws;  ws += (size_t)S_LEN * (NHQ * HDIM) * sizeof(float);
  float* Kpre = (float*)ws;  ws += (size_t)S_LEN * (NKV * HDIM) * sizeof(float);
  float* Vpre = (float*)ws;  ws += (size_t)S_LEN * (NKV * HDIM) * sizeof(float);
  __bf16* Qr = (__bf16*)ws;  ws += (size_t)S_LEN * (NHQ * HDIM) * sizeof(__bf16);
  __bf16* Kr = (__bf16*)ws;  ws += (size_t)S_LEN * (NKV * HDIM) * sizeof(__bf16);
  __bf16* Vb = (__bf16*)ws;  ws += (size_t)S_LEN * (NKV * HDIM) * sizeof(__bf16);
  float* attnO = Qpre;  // Qpre is dead after the RoPE kernel; reuse it

  // QKV projections (C = x * W^T)
  gemm_xwT_bf16<4096, 4096><<<dim3(32, 32), 256, 0, stream>>>(x, wq, Qpre);
  gemm_xwT_bf16<1024, 4096><<<dim3(8, 32), 256, 0, stream>>>(x, wk, Kpre);
  gemm_xwT_bf16<1024, 4096><<<dim3(8, 32), 256, 0, stream>>>(x, wv, Vpre);

  // RoPE + QK RMSNorm + temp scale, cast to bf16
  rope_rms_cast_kernel<<<dim3((S_LEN * NHQ) / 8), 256, 0, stream>>>(Qpre, Qr, NHQ, 1);
  rope_rms_cast_kernel<<<dim3((S_LEN * NKV) / 8), 256, 0, stream>>>(Kpre, Kr, NKV, 0);
  cast_bf16_kernel<<<dim3((S_LEN * NKV * HDIM / 4) / 256), 256, 0, stream>>>(
      Vpre, Vb, S_LEN * NKV * HDIM / 4);

  // chunked-causal attention
  attn_chunked_kernel<<<dim3(8, 32, 4), 256, 0, stream>>>(Qr, Kr, Vb, attnO);

  // output projection
  gemm_xwT_bf16<4096, 4096><<<dim3(32, 32), 256, 0, stream>>>(attnO, wo, out);
}
